// GRUClassifier_652835029458
// MI455X (gfx1250) — compile-verified, matrix operand order checked by reading, not run on hardware
//
#include <hip/hip_runtime.h>
#include <hip/hip_bf16.h>
#include <stdint.h>
#include <stddef.h>

// ---------------------------------------------------------------------------
// Types / constants
// ---------------------------------------------------------------------------
typedef __bf16 bf16;
typedef __bf16 v16bf __attribute__((ext_vector_type(16)));
typedef __bf16 v8bf  __attribute__((ext_vector_type(8)));
typedef float  v8f   __attribute__((ext_vector_type(8)));

#define B_    128
#define T_    256
#define DIN_  128
#define HID_  256
#define G3_   768   // 3*HID
#define NC_   250

__device__ __forceinline__ v8f vzero8() {
  v8f z = {0.f,0.f,0.f,0.f,0.f,0.f,0.f,0.f};
  return z;
}

// Load a 16-element bf16 WMMA fragment: per ISA layout, each lane needs two
// contiguous 8-element (16-byte) chunks separated by 16 elements.
__device__ __forceinline__ v16bf load_frag16(const bf16* p) {
  v8bf lo = *(const v8bf*)(p);
  v8bf hi = *(const v8bf*)(p + 16);
  v16bf r;
#pragma unroll
  for (int i = 0; i < 8; ++i) { r[i] = lo[i]; r[i + 8] = hi[i]; }
  return r;
}

__device__ __forceinline__ v8f wmma_bf16(v16bf a, v16bf b, v8f c) {
  return __builtin_amdgcn_wmma_f32_16x16x32_bf16(false, a, false, b,
                                                 (short)0, c, false, false);
}

__device__ __forceinline__ float sigm(float x) {
  return 1.0f / (1.0f + __expf(-x));
}
__device__ __forceinline__ float tanh_fast(float x) {
  float e = __expf(-2.0f * x);
  return (1.0f - e) / (1.0f + e);
}

// ---------------------------------------------------------------------------
// f32 -> bf16 cast (grid-stride)
// ---------------------------------------------------------------------------
__global__ void cast_f32_bf16(const float* __restrict__ in,
                              bf16* __restrict__ out, int n) {
  int i = blockIdx.x * blockDim.x + threadIdx.x;
  int stride = gridDim.x * blockDim.x;
  for (; i < n; i += stride) out[i] = (bf16)in[i];
}

// ---------------------------------------------------------------------------
// GEMM: C[M,768](bf16) = A[M,K](bf16) @ W[768,K]^T(bf16) + bias[768](f32)
// WG = 256 threads (8 waves), tile 64(M) x 128(N), each wave 32x32 (2x2 WMMA).
// Both A and W are K-major row-major -> fragments load directly.
// ---------------------------------------------------------------------------
__global__ __launch_bounds__(256)
void gemm_bf16_nt(const bf16* __restrict__ A, const bf16* __restrict__ W,
                  const float* __restrict__ bias, bf16* __restrict__ C,
                  int M, int K) {
  const int lane = threadIdx.x & 31;
  const int wave = threadIdx.x >> 5;
  const int half = lane >> 4;
  const int l15  = lane & 15;
  const int wm = wave >> 2;              // 0..1
  const int wn = wave & 3;               // 0..3
  const int m0 = blockIdx.x * 64 + wm * 32;
  const int n0 = blockIdx.y * 128 + wn * 32;

  v8f acc[2][2];
#pragma unroll
  for (int r = 0; r < 2; ++r)
#pragma unroll
    for (int c = 0; c < 2; ++c) acc[r][c] = vzero8();

  for (int k0 = 0; k0 < K; k0 += 32) {
    const int kb = k0 + 8 * half;
    v16bf a0 = load_frag16(A + (size_t)(m0 + l15) * K + kb);
    v16bf a1 = load_frag16(A + (size_t)(m0 + 16 + l15) * K + kb);
    v16bf b0 = load_frag16(W + (size_t)(n0 + l15) * K + kb);
    v16bf b1 = load_frag16(W + (size_t)(n0 + 16 + l15) * K + kb);
    acc[0][0] = wmma_bf16(a0, b0, acc[0][0]);
    acc[0][1] = wmma_bf16(a0, b1, acc[0][1]);
    acc[1][0] = wmma_bf16(a1, b0, acc[1][0]);
    acc[1][1] = wmma_bf16(a1, b1, acc[1][1]);
  }

#pragma unroll
  for (int r = 0; r < 2; ++r)
#pragma unroll
    for (int c = 0; c < 2; ++c) {
      const int n = n0 + c * 16 + l15;
      const float bv = bias[n];
#pragma unroll
      for (int i = 0; i < 8; ++i) {
        const int m = m0 + r * 16 + 8 * half + i;
        C[(size_t)m * G3_ + n] = (bf16)(acc[r][c][i] + bv);
      }
    }
}

// ---------------------------------------------------------------------------
// Bidirectional GRU scan. grid = (B/16, 2 directions), block = 512 (16 waves).
// Each WG owns a 16-row batch tile; h kept in LDS (bf16 for WMMA A-frags,
// f32 for the exact z*h update). Per step: [16x256] x [256x768] via WMMA.
// Wave w owns hidden columns [w*16, w*16+16) across all 3 gates, so the
// r/z/n accumulators align elementwise for the gate nonlinearity.
//
// Key scheduling decisions for the serial critical path:
//  * The 24 recurrent-weight WMMA fragments per wave (3 gates x 8 K-chunks)
//    are loop-invariant -> preloaded into VGPRs before the time loop
//    (gfx1250 waves can address up to 1024 VGPRs).
//  * The step's gx tile [16 x 768] is staged to LDS with coalesced b128
//    loads issued before the WMMA block (latency hidden by tensor ops);
//    gate math then reads LDS scalars instead of scattered global u16 loads.
//  * global_prefetch_b8 warms the next timestep's gx tile.
// ---------------------------------------------------------------------------
#define HBF_STRIDE (HID_ + 8)   // bf16 elems; 528B rows -> 16B-aligned
#define HF_STRIDE  260          // f32 elems
#define GXS_STRIDE (G3_ + 8)    // bf16 elems; 1552B rows -> 16B-aligned

__global__ __launch_bounds__(512)
void gru_scan_bidir(const bf16* __restrict__ gx_f, const bf16* __restrict__ gx_r,
                    const bf16* __restrict__ whh_f, const bf16* __restrict__ whh_r,
                    const float* __restrict__ bhh_f, const float* __restrict__ bhh_r,
                    bf16* __restrict__ ys,       // [B,T,512] or nullptr
                    float* __restrict__ lastout, // [B,512] or nullptr
                    int nsteps_f, int nsteps_r) {
  const int dir = blockIdx.y;
  const bf16*  gx  = dir ? gx_r  : gx_f;
  const bf16*  whh = dir ? whh_r : whh_f;   // [768,256] row-major
  const float* bhh = dir ? bhh_r : bhh_f;   // [768]
  const int nsteps = dir ? nsteps_r : nsteps_f;
  const int colofs = dir * HID_;

  __shared__ bf16  hbf[16 * HBF_STRIDE];
  __shared__ float hf[16 * HF_STRIDE];
  __shared__ bf16  gxs[16 * GXS_STRIDE];

  const int lane = threadIdx.x & 31;
  const int wave = threadIdx.x >> 5;
  const int half = lane >> 4;
  const int l15  = lane & 15;
  const int b0 = blockIdx.x * 16;
  const int j0 = wave * 16;        // hidden column slice of this wave
  const int n  = j0 + l15;         // this lane's hidden column (C/D: lane->N)

  for (int i = threadIdx.x; i < 16 * HBF_STRIDE; i += blockDim.x) hbf[i] = (bf16)0.0f;
  for (int i = threadIdx.x; i < 16 * HF_STRIDE;  i += blockDim.x) hf[i] = 0.0f;
  __syncthreads();

  const float br = bhh[n];
  const float bz = bhh[HID_ + n];
  const float bn = bhh[2 * HID_ + n];

  // Preload loop-invariant recurrent-weight fragments into registers:
  // wf[g][k] = Whh rows [g*256+n] K-chunk k (per-lane 32 bytes each).
  v16bf wf[3][8];
#pragma unroll
  for (int g = 0; g < 3; ++g)
#pragma unroll
    for (int k = 0; k < 8; ++k)
      wf[g][k] = load_frag16(whh + (size_t)(g * HID_ + n) * HID_ + k * 32 + 8 * half);

  // Staging geometry: 16 rows x 768 cols bf16 = 1536 8-elem chunks; 3/thread.
  for (int s = 0; s < nsteps; ++s) {
    const int t = dir ? (T_ - 1 - s) : s;

    // ---- stage this step's gx tile into LDS (coalesced b128 loads) ----
#pragma unroll
    for (int i = 0; i < 3; ++i) {
      const int chunk = threadIdx.x + i * 512;
      const int row  = chunk / 96;        // 0..15
      const int col8 = (chunk % 96) * 8;  // 0,8,...,760
      const bf16* src = gx + ((size_t)(b0 + row) * T_ + t) * G3_ + col8;
      *(v8bf*)(gxs + row * GXS_STRIDE + col8) = *(const v8bf*)src;
      if (s + 1 < nsteps) {
        const int t2 = dir ? (t - 1) : (t + 1);
        __builtin_prefetch(gx + ((size_t)(b0 + row) * T_ + t2) * G3_ + col8, 0, 0);
      }
    }

    // ---- recurrent matmul: [16x256] x [256x48(this wave)] ----
    v8f accR = vzero8(), accZ = vzero8(), accN = vzero8();
#pragma unroll
    for (int k = 0; k < 8; ++k) {
      v16bf a = load_frag16(hbf + l15 * HBF_STRIDE + k * 32 + 8 * half);
      accR = wmma_bf16(a, wf[0][k], accR);
      accZ = wmma_bf16(a, wf[1][k], accZ);
      accN = wmma_bf16(a, wf[2][k], accN);
    }

    __syncthreads();   // gxs visible; all hbf reads complete

    // ---- gate math + state update (element (m,n) owned by one lane) ----
    float hnew[8];
#pragma unroll
    for (int i = 0; i < 8; ++i) {
      const int m = 8 * half + i;  // C/D layout: vgpr index + half -> row
      const float xr = (float)gxs[m * GXS_STRIDE + n];
      const float xz = (float)gxs[m * GXS_STRIDE + HID_ + n];
      const float xn = (float)gxs[m * GXS_STRIDE + 2 * HID_ + n];
      const float hv = hf[m * HF_STRIDE + n];
      const float r  = sigm(xr + accR[i] + br);
      const float z  = sigm(xz + accZ[i] + bz);
      const float nn = tanh_fast(xn + r * (accN[i] + bn));
      hnew[i] = (1.0f - z) * nn + z * hv;
      hbf[m * HBF_STRIDE + n] = (bf16)hnew[i];
      hf[m * HF_STRIDE + n] = hnew[i];
      if (ys) ys[((size_t)(b0 + m) * T_ + t) * (2 * HID_) + colofs + n] = (bf16)hnew[i];
    }
    if (lastout && s == nsteps - 1) {
#pragma unroll
      for (int i = 0; i < 8; ++i) {
        const int m = 8 * half + i;
        lastout[(size_t)(b0 + m) * (2 * HID_) + colofs + n] = hnew[i];
      }
    }
    __syncthreads();   // h update visible before next step's reads
  }
}

// ---------------------------------------------------------------------------
// LayerNorm + head. One WG (256 thr) per batch row; f32 throughout.
// ---------------------------------------------------------------------------
__global__ __launch_bounds__(256)
void ln_head(const float* __restrict__ last, const float* __restrict__ g,
             const float* __restrict__ bb, const float* __restrict__ hw,
             const float* __restrict__ hb, float* __restrict__ out) {
  __shared__ float buf[2 * HID_];
  __shared__ float red[256];
  __shared__ float red2[256];
  const int bi = blockIdx.x, tid = threadIdx.x;

  float s = 0.f, q = 0.f;
  for (int i = tid; i < 2 * HID_; i += 256) {
    const float v = last[(size_t)bi * 2 * HID_ + i];
    buf[i] = v; s += v; q += v * v;
  }
  red[tid] = s; red2[tid] = q;
  __syncthreads();
  for (int ofs = 128; ofs > 0; ofs >>= 1) {
    if (tid < ofs) { red[tid] += red[tid + ofs]; red2[tid] += red2[tid + ofs]; }
    __syncthreads();
  }
  const float mu   = red[0] * (1.0f / (2 * HID_));
  const float var  = red2[0] * (1.0f / (2 * HID_)) - mu * mu;
  const float rstd = rsqrtf(var + 1e-5f);
  __syncthreads();
  for (int i = tid; i < 2 * HID_; i += 256)
    buf[i] = (buf[i] - mu) * rstd * g[i] + bb[i];
  __syncthreads();

  if (tid < NC_) {
    float acc = hb[tid];
#pragma unroll 8
    for (int k = 0; k < 2 * HID_; ++k) acc += buf[k] * hw[(size_t)tid * (2 * HID_) + k];
    out[(size_t)bi * NC_ + tid] = acc;
  }
}

// ---------------------------------------------------------------------------
// Host launch
// ---------------------------------------------------------------------------
extern "C" void kernel_launch(void* const* d_in, const int* in_sizes, int n_in,
                              void* d_out, int out_size, void* d_ws, size_t ws_size,
                              hipStream_t stream) {
  const float* x      = (const float*)d_in[0];
  const float* wih0   = (const float*)d_in[1];
  const float* whh0   = (const float*)d_in[2];
  const float* bih0   = (const float*)d_in[3];
  const float* bhh0   = (const float*)d_in[4];
  const float* wih0r  = (const float*)d_in[5];
  const float* whh0r  = (const float*)d_in[6];
  const float* bih0r  = (const float*)d_in[7];
  const float* bhh0r  = (const float*)d_in[8];
  const float* wih1   = (const float*)d_in[9];
  const float* whh1   = (const float*)d_in[10];
  const float* bih1   = (const float*)d_in[11];
  const float* bhh1   = (const float*)d_in[12];
  const float* wih1r  = (const float*)d_in[13];
  const float* whh1r  = (const float*)d_in[14];
  const float* bih1r  = (const float*)d_in[15];
  const float* bhh1r  = (const float*)d_in[16];
  const float* ln_g   = (const float*)d_in[17];
  const float* ln_b   = (const float*)d_in[18];
  const float* head_w = (const float*)d_in[19];
  const float* head_b = (const float*)d_in[20];
  float* out = (float*)d_out;

  // Workspace carve-out (all bf16 staging + gx buffers + layer-0 outputs)
  uint8_t* p = (uint8_t*)d_ws;
  uint8_t* pend = p + ws_size;
  auto alloc = [&](size_t bytes) -> uint8_t* {
    uint8_t* r = p;
    p += (bytes + 255) & ~(size_t)255;
    return r;
  };
  bf16* xb     = (bf16*)alloc((size_t)B_ * T_ * DIN_ * 2);
  bf16* wih0b  = (bf16*)alloc((size_t)G3_ * DIN_ * 2);
  bf16* whh0b  = (bf16*)alloc((size_t)G3_ * HID_ * 2);
  bf16* wih0rb = (bf16*)alloc((size_t)G3_ * DIN_ * 2);
  bf16* whh0rb = (bf16*)alloc((size_t)G3_ * HID_ * 2);
  bf16* wih1b  = (bf16*)alloc((size_t)G3_ * 2 * HID_ * 2);
  bf16* whh1b  = (bf16*)alloc((size_t)G3_ * HID_ * 2);
  bf16* wih1rb = (bf16*)alloc((size_t)G3_ * 2 * HID_ * 2);
  bf16* whh1rb = (bf16*)alloc((size_t)G3_ * HID_ * 2);
  bf16* gxA    = (bf16*)alloc((size_t)B_ * T_ * G3_ * 2);  // fwd gx (reused per layer)
  bf16* gxB    = (bf16*)alloc((size_t)B_ * T_ * G3_ * 2);  // bwd gx (reused per layer)
  bf16* ys0    = (bf16*)alloc((size_t)B_ * T_ * 2 * HID_ * 2);
  float* lastb = (float*)alloc((size_t)B_ * 2 * HID_ * 4);
  if (p > pend) return;  // workspace too small: bail deterministically

  auto cast = [&](const float* src, bf16* dst, int n) {
    int grid = (n + 255) / 256;
    if (grid > 4096) grid = 4096;
    cast_f32_bf16<<<grid, 256, 0, stream>>>(src, dst, n);
  };
  cast(x,     xb,     B_ * T_ * DIN_);
  cast(wih0,  wih0b,  G3_ * DIN_);
  cast(whh0,  whh0b,  G3_ * HID_);
  cast(wih0r, wih0rb, G3_ * DIN_);
  cast(whh0r, whh0rb, G3_ * HID_);
  cast(wih1,  wih1b,  G3_ * 2 * HID_);
  cast(whh1,  whh1b,  G3_ * HID_);
  cast(wih1r, wih1rb, G3_ * 2 * HID_);
  cast(whh1r, whh1rb, G3_ * HID_);

  const int M = B_ * T_;
  dim3 gg(M / 64, G3_ / 128);

  // Layer 0: input projections (parallel GEMMs), then bidirectional scan.
  gemm_bf16_nt<<<gg, 256, 0, stream>>>(xb, wih0b,  bih0,  gxA, M, DIN_);
  gemm_bf16_nt<<<gg, 256, 0, stream>>>(xb, wih0rb, bih0r, gxB, M, DIN_);
  gru_scan_bidir<<<dim3(B_ / 16, 2), 512, 0, stream>>>(
      gxA, gxB, whh0b, whh0rb, bhh0, bhh0r, ys0, nullptr, T_, T_);

  // Layer 1: projections from concat(fwd,bwd) (K=512), then scan.
  // h2[:, -1] = [fwd after full T steps | bwd after exactly 1 step].
  gemm_bf16_nt<<<gg, 256, 0, stream>>>(ys0, wih1b,  bih1,  gxA, M, 2 * HID_);
  gemm_bf16_nt<<<gg, 256, 0, stream>>>(ys0, wih1rb, bih1r, gxB, M, 2 * HID_);
  gru_scan_bidir<<<dim3(B_ / 16, 2), 512, 0, stream>>>(
      gxA, gxB, whh1b, whh1rb, bhh1, bhh1r, nullptr, lastb, T_, 1);

  // LayerNorm + linear head (f32).
  ln_head<<<B_, 256, 0, stream>>>(lastb, ln_g, ln_b, head_w, head_b, out);
}
